// CUSTOM_RNN_LAYER_58514634440782
// MI455X (gfx1250) — matrix operand order repaired and checked by reading, not compiled, not run on hardware
//
#include <hip/hip_runtime.h>
#include <hip/hip_bf16.h>
#include <math.h>
#include <stdint.h>

// Problem constants (match reference)
#define BB   256
#define TT   512
#define II   128
#define HH   256
#define H2   512   // 2*H
#define H3   768   // 3*H
#define TEMP_INV 10.0f

typedef __attribute__((ext_vector_type(16))) _Float16 v16h;
typedef __attribute__((ext_vector_type(8)))  float    v8f;

__device__ __forceinline__ float sigmoidf_(float x) { return 1.0f / (1.0f + expf(-x)); }

// ---------------------------------------------------------------- utilities
__global__ void cvt_f32_f16(const float* __restrict__ s, _Float16* __restrict__ d, int n) {
  int i = blockIdx.x * blockDim.x + threadIdx.x;
  if (i < n) d[i] = (_Float16)s[i];
}

__global__ void zero_f32(float* __restrict__ p, int n) {
  int i = blockIdx.x * blockDim.x + threadIdx.x;
  if (i < n) p[i] = 0.0f;
}

// com[b, 0:256] = c_state[b,:]; com[b, 256:512] = c_in(t)[b,:] (from c_concat in d_out)
__global__ void pack_com(const float* __restrict__ c_state,
                         const float* __restrict__ c_concat,
                         float* __restrict__ com, int t) {
  int idx = blockIdx.x * blockDim.x + threadIdx.x;   // [0, BB*H2)
  int b = idx >> 9, col = idx & 511;
  float v;
  if (col < HH) v = c_state[(b << 8) + col];
  else          v = c_concat[((size_t)b * TT + t) * HH + (col - HH)];
  com[idx] = v;
}

// ------------------------------------------------------------------ WMMA GEMM
// Out[M,N] = act( A[M,K](f32) @ W[N,K](f16)^T + bias[N] )
// block = 128 threads (4 wave32), block tile = 32(M) x 64(N), K-step = 32.
// Each wave: one 16-col N-slice, two 16-row M-slices -> 2 WMMAs / K-tile
// sharing a single B fragment. B tiles move via the gfx1250 async
// global->LDS path (ASYNCcnt); A tiles are f32->f16 converted in registers.
// Requires M%32==0, N%64==0, K%32==0 (true for all shapes used here).
// act: 0 = none, 1 = relu, 2 = tanh
__global__ __launch_bounds__(128)
void gemm_wmma(const float* __restrict__ A, int lda,
               const _Float16* __restrict__ W,
               const float* __restrict__ bias,
               float* __restrict__ out, int ldo,
               int K, int act)
{
  __shared__ _Float16 ldsA[32][32];        // A tile, row-major (m, k)
  __shared__ _Float16 ldsB[4][16][32];     // per-wave W tile, (n, k)

  const int tid  = threadIdx.x;
  const int wave = tid >> 5;
  const int lane = tid & 31;
  const int m0 = blockIdx.x * 32;
  const int n0 = blockIdx.y * 64 + wave * 16;
  const int hi = lane >> 4;
  const int lr = lane & 15;

  // A-tile fill mapping: 128 threads cover 16 rows x 8 float4-chunks,
  // done twice for the 32 rows.
  const int ar = tid >> 3;           // 0..15
  const int ac = (tid & 7) << 2;     // 0,4,...,28

  const unsigned bBase = (unsigned)(uintptr_t)&ldsB[wave][0][0];
  const char* Wbytes = (const char*)W;

  v8f acc0 = {}, acc1 = {};
  const int nK = K >> 5;

  for (int kt = 0; kt < nK; ++kt) {
    const int k0 = kt << 5;

    // ---- B tile: async global->LDS copy, 4 x b64 per wave (1KB tile).
    // Each lane moves 8 bytes per op; rows n0..n0+15, cols k0..k0+31.
    #pragma unroll
    for (int i = 0; i < 4; ++i) {
      int chunk = (i << 5) + lane;             // 0..127 (8-byte chunks)
      int r  = chunk >> 3;                     // 0..15  (n row)
      int c8 = chunk & 7;                      // 8-byte column chunk
      const char* gp = Wbytes + (((size_t)(n0 + r) * K + k0) << 1) + (c8 << 3);
      unsigned loff = bBase + (r << 6) + (c8 << 3);
      asm volatile("global_load_async_to_lds_b64 %0, %1, off"
                   :: "v"(loff), "v"(gp) : "memory");
    }

    // ---- A tile: two float4 loads -> f16 pack -> b64 LDS stores
    float4 a0 = *(const float4*)&A[(size_t)(m0 + ar) * lda + (k0 + ac)];
    float4 a1 = *(const float4*)&A[(size_t)(m0 + 16 + ar) * lda + (k0 + ac)];
    union Pack { _Float16 h[4]; uint2 u; };
    Pack p0, p1;
    p0.h[0] = (_Float16)a0.x; p0.h[1] = (_Float16)a0.y;
    p0.h[2] = (_Float16)a0.z; p0.h[3] = (_Float16)a0.w;
    p1.h[0] = (_Float16)a1.x; p1.h[1] = (_Float16)a1.y;
    p1.h[2] = (_Float16)a1.z; p1.h[3] = (_Float16)a1.w;
    *(uint2*)&ldsA[ar][ac]      = p0.u;
    *(uint2*)&ldsA[16 + ar][ac] = p1.u;

    // Prefetch next K-tile (global_prefetch_b8)
    if (kt + 1 < nK) {
      __builtin_prefetch(&A[(size_t)(m0 + ar) * lda + (k0 + 32)], 0, 1);
      __builtin_prefetch(&W[(size_t)(n0 + lr) * K + (k0 + 32)], 0, 1);
    }

    asm volatile("s_wait_asynccnt 0x0" ::: "memory");  // B tile landed in LDS
    __syncthreads();                                   // A tile visible

    // Assemble fragments per CDNA5 16-bit A/B VGPR layout (wave32).
    // A 16x32: lane m=lr; vgpr v<4 -> K=2v(+8 hi), v>=4 -> K=16+2(v-4)(+8 hi)
    // B 32x16: lane n=lr; vgpr v   -> same K as A
    v16h av0, av1, bv;
    #pragma unroll
    for (int v = 0; v < 8; ++v) {
      int ka = (v < 4) ? (2 * v + 8 * hi) : (16 + 2 * (v - 4) + 8 * hi);
      av0[2 * v]     = ldsA[lr][ka];
      av0[2 * v + 1] = ldsA[lr][ka + 1];
      av1[2 * v]     = ldsA[16 + lr][ka];
      av1[2 * v + 1] = ldsA[16 + lr][ka + 1];
      int kb = ka;
      bv[2 * v]     = ldsB[wave][lr][kb];
      bv[2 * v + 1] = ldsB[wave][lr][kb + 1];
    }
    acc0 = __builtin_amdgcn_wmma_f32_16x16x32_f16(
        false, av0, false, bv, (short)0, acc0, false, false);
    acc1 = __builtin_amdgcn_wmma_f32_16x16x32_f16(
        false, av1, false, bv, (short)0, acc1, false, false);
    __syncthreads();
  }

  // C/D layout: lane lr = column n; vgpr r -> row r (+8 for hi lanes)
  #pragma unroll
  for (int r = 0; r < 8; ++r) {
    int nn = n0 + lr;
    float bsv = bias[nn];
    float v0 = acc0[r] + bsv;
    float v1 = acc1[r] + bsv;
    if (act == 1)      { v0 = fmaxf(v0, 0.0f); v1 = fmaxf(v1, 0.0f); }
    else if (act == 2) { v0 = tanhf(v0);       v1 = tanhf(v1); }
    out[(size_t)(m0 + r + 8 * hi) * ldo + nn]      = v0;
    out[(size_t)(m0 + 16 + r + 8 * hi) * ldo + nn] = v1;
  }
}

// -------------------------------------------------- per-step fused update
// One block per batch row b; 256 threads (one per h column).
// logit = a3[b,:] . W4 + b4 ; alpha = sigmoid((logit+logistic)/TEMP)
// GRU combine + state updates + h_concat write.
__global__ __launch_bounds__(256)
void step_update(const float* __restrict__ a3,
                 const float* __restrict__ W4, const float* __restrict__ b4,
                 const float* __restrict__ noise,
                 const float* __restrict__ gi, const float* __restrict__ gh,
                 const float* __restrict__ c_concat, float* __restrict__ h_concat,
                 float* __restrict__ c_state, float* __restrict__ h_state,
                 float* __restrict__ n_state, int t)
{
  __shared__ float red[256];
  __shared__ float s_alpha;
  const int b = blockIdx.x;
  const int j = threadIdx.x;

  const float* a3b = a3 + (size_t)b * H2;
  red[j] = a3b[j] * W4[j] + a3b[j + HH] * W4[j + HH];
  __syncthreads();
  #pragma unroll
  for (int s = 128; s > 0; s >>= 1) {
    if (j < s) red[j] += red[j + s];
    __syncthreads();
  }
  if (j == 0) {
    float logit = red[0] + b4[0];
    float u = noise[(size_t)t * BB + b];           // noise is [T,B,1]
    float logistic = logf(u) - log1pf(-u);
    s_alpha = sigmoidf_((logit + logistic) * TEMP_INV);
  }
  __syncthreads();
  const float alpha = s_alpha;
  const float nold  = n_state[b];
  __syncthreads();   // all reads of n_state done before lane 0 rewrites it

  const size_t gb = (size_t)b * H3;
  float r  = sigmoidf_(gi[gb + j]          + gh[gb + j]);
  float z  = sigmoidf_(gi[gb + HH + j]     + gh[gb + HH + j]);
  float ng = tanhf    (gi[gb + 2 * HH + j] + r * gh[gb + 2 * HH + j]);

  const size_t hb = (size_t)b * HH + j;
  float hold = h_state[hb];
  float gru  = (1.0f - z) * ng + z * hold;
  float hnew = hold * (1.0f - alpha) + alpha * gru;
  h_state[hb] = hnew;
  h_concat[((size_t)b * TT + t) * HH + j] = hnew;

  float cin  = c_concat[((size_t)b * TT + t) * HH + j];
  float nnew = nold * (1.0f - alpha) + 1.0f;
  float cold = c_state[hb];
  c_state[hb] = (cold * nold * (1.0f - alpha) + cin) / nnew;
  if (j == 0) n_state[b] = nnew;
}

// h_final = gru_cell(c_T, h_T) from precomputed gi/gh
__global__ __launch_bounds__(256)
void final_gru(const float* __restrict__ gi, const float* __restrict__ gh,
               const float* __restrict__ h_state, float* __restrict__ h_final)
{
  const int b = blockIdx.x;
  const int j = threadIdx.x;
  const size_t gb = (size_t)b * H3;
  float r  = sigmoidf_(gi[gb + j]          + gh[gb + j]);
  float z  = sigmoidf_(gi[gb + HH + j]     + gh[gb + HH + j]);
  float ng = tanhf    (gi[gb + 2 * HH + j] + r * gh[gb + 2 * HH + j]);
  float h  = h_state[(size_t)b * HH + j];
  h_final[(size_t)b * HH + j] = (1.0f - z) * ng + z * h;
}

// ------------------------------------------------------------------- launch
extern "C" void kernel_launch(void* const* d_in, const int* in_sizes, int n_in,
                              void* d_out, int out_size, void* d_ws, size_t ws_size,
                              hipStream_t stream) {
  (void)in_sizes; (void)n_in; (void)out_size; (void)ws_size;

  const float* input = (const float*)d_in[0];   // [B,T,I]
  const float* noise = (const float*)d_in[1];   // [T,B,1]
  const float* W_cin = (const float*)d_in[2];   // [H,I]
  const float* b_cin = (const float*)d_in[3];
  const float* W1    = (const float*)d_in[4];   // [2H,2H]
  const float* b1    = (const float*)d_in[5];
  const float* W2    = (const float*)d_in[6];
  const float* b2    = (const float*)d_in[7];
  const float* W3    = (const float*)d_in[8];
  const float* b3    = (const float*)d_in[9];
  const float* W4    = (const float*)d_in[10];  // [1,2H]
  const float* b4    = (const float*)d_in[11];
  const float* Wih   = (const float*)d_in[12];  // [3H,H]
  const float* Whh   = (const float*)d_in[13];  // [3H,H]
  const float* bih   = (const float*)d_in[14];
  const float* bhh   = (const float*)d_in[15];

  float* c_concat = (float*)d_out;                          // [B,T,H]
  float* h_concat = c_concat + (size_t)BB * TT * HH;        // [B,T,H]
  float* h_final  = h_concat + (size_t)BB * TT * HH;        // [B,H]

  // workspace carve-out (256B aligned)
  size_t off = 0;
  auto alloc = [&](size_t bytes) -> void* {
    void* p = (char*)d_ws + off;
    off += (bytes + 255) & ~(size_t)255;
    return p;
  };
  _Float16* Wc_h  = (_Float16*)alloc((size_t)HH * II * sizeof(_Float16));
  _Float16* W1_h  = (_Float16*)alloc((size_t)H2 * H2 * sizeof(_Float16));
  _Float16* W2_h  = (_Float16*)alloc((size_t)H2 * H2 * sizeof(_Float16));
  _Float16* W3_h  = (_Float16*)alloc((size_t)H2 * H2 * sizeof(_Float16));
  _Float16* Wih_h = (_Float16*)alloc((size_t)H3 * HH * sizeof(_Float16));
  _Float16* Whh_h = (_Float16*)alloc((size_t)H3 * HH * sizeof(_Float16));
  float* com     = (float*)alloc((size_t)BB * H2 * sizeof(float));
  float* a1      = (float*)alloc((size_t)BB * H2 * sizeof(float));
  float* a2      = (float*)alloc((size_t)BB * H2 * sizeof(float));
  float* a3      = (float*)alloc((size_t)BB * H2 * sizeof(float));
  float* gi      = (float*)alloc((size_t)BB * H3 * sizeof(float));
  float* gh      = (float*)alloc((size_t)BB * H3 * sizeof(float));
  float* c_state = (float*)alloc((size_t)BB * HH * sizeof(float));
  float* h_state = (float*)alloc((size_t)BB * HH * sizeof(float));
  float* n_state = (float*)alloc((size_t)BB * sizeof(float));

  // ---- weight converts (f32 -> f16), fresh every call (deterministic)
  auto cvt = [&](const float* s, _Float16* d, int n) {
    cvt_f32_f16<<<(n + 255) / 256, 256, 0, stream>>>(s, d, n);
  };
  cvt(W_cin, Wc_h,  HH * II);
  cvt(W1,    W1_h,  H2 * H2);
  cvt(W2,    W2_h,  H2 * H2);
  cvt(W3,    W3_h,  H2 * H2);
  cvt(Wih,   Wih_h, H3 * HH);
  cvt(Whh,   Whh_h, H3 * HH);

  // ---- zero initial states
  zero_f32<<<(BB * HH + 255) / 256, 256, 0, stream>>>(c_state, BB * HH);
  zero_f32<<<(BB * HH + 255) / 256, 256, 0, stream>>>(h_state, BB * HH);
  zero_f32<<<(BB + 255) / 256, 256, 0, stream>>>(n_state, BB);

  // ---- phase 1: c_in for all (b,t) in one big WMMA GEMM, tanh fused,
  //      written directly into the c_concat output region.
  {
    dim3 grid((BB * TT) / 32, HH / 64);
    gemm_wmma<<<grid, 128, 0, stream>>>(input, II, Wc_h, b_cin,
                                        c_concat, HH, II, /*act=*/2);
  }

  // ---- phase 2: sequential recurrence over T
  dim3 gMLP(BB / 32, H2 / 64);   // 8 x 8
  dim3 gGRU(BB / 32, H3 / 64);   // 8 x 12
  for (int t = 0; t < TT; ++t) {
    pack_com<<<(BB * H2) / 256, 256, 0, stream>>>(c_state, c_concat, com, t);
    gemm_wmma<<<gMLP, 128, 0, stream>>>(com, H2, W1_h, b1, a1, H2, H2, 1);
    gemm_wmma<<<gMLP, 128, 0, stream>>>(a1,  H2, W2_h, b2, a2, H2, H2, 1);
    gemm_wmma<<<gMLP, 128, 0, stream>>>(a2,  H2, W3_h, b3, a3, H2, H2, 1);
    // GRU gate GEMMs read pre-update c_state / h_state (stream order).
    gemm_wmma<<<gGRU, 128, 0, stream>>>(c_state, HH, Wih_h, bih, gi, H3, HH, 0);
    gemm_wmma<<<gGRU, 128, 0, stream>>>(h_state, HH, Whh_h, bhh, gh, H3, HH, 0);
    step_update<<<BB, 256, 0, stream>>>(a3, W4, b4, noise, gi, gh,
                                        c_concat, h_concat,
                                        c_state, h_state, n_state, t);
  }

  // ---- final h: gru_cell(c_T, h_T)
  gemm_wmma<<<gGRU, 128, 0, stream>>>(c_state, HH, Wih_h, bih, gi, H3, HH, 0);
  gemm_wmma<<<gGRU, 128, 0, stream>>>(h_state, HH, Whh_h, bhh, gh, H3, HH, 0);
  final_gru<<<BB, 256, 0, stream>>>(gi, gh, h_state, h_final);
}
